// LRU_60181081752357
// MI455X (gfx1250) — compile-verified, matrix operand order checked by reading, not run on hardware
//
#include <hip/hip_runtime.h>
#include <hip/hip_bf16.h>
#include <math.h>

// LRU single-step:  y = Re((0.5*lam_mix(h_old) + X @ (gamma*B)^T) @ C^T) + D*X
// f16-WMMA GEMMs; all f32->f16 conversion hoisted into a bandwidth-bound prep
// pass so the GEMM inner loops are pure {b128 load + v_wmma} streams.

typedef __attribute__((ext_vector_type(16))) _Float16 v16h;
typedef __attribute__((ext_vector_type(8)))  _Float16 v8h;
typedef __attribute__((ext_vector_type(8)))  float    v8f;

#define N_HID   2048   // N
#define TWO_N   4096   // 2N
#define DMODEL  2048
#define NBATCH  4096

// ---------------------------------------------------------------------------
// f16 A (16x32 MxK) / B (32x16 KxN) WMMA fragments: lane L (r = L&15,
// hf = L>>4) holds two contiguous 8-element K-chunks of row r:
//   K = k0 + hf*8 + [0..7]   and   K = k0 + 16 + hf*8 + [0..7]
// => two 128-bit loads per lane per fragment.
// ---------------------------------------------------------------------------
__device__ __forceinline__ v16h load_frag_h16(const _Float16* __restrict__ p) {
  v8h lo = *reinterpret_cast<const v8h*>(p);        // k .. k+7
  v8h hi = *reinterpret_cast<const v8h*>(p + 16);   // k+16 .. k+23
  v16h o;
#pragma unroll
  for (int i = 0; i < 8; ++i) { o[i] = lo[i]; o[8 + i] = hi[i]; }
  return o;
}

#define WMMA_F16(A, B, C) \
  __builtin_amdgcn_wmma_f32_16x16x32_f16(false, (A), false, (B), (short)0, (C), false, false)

// ---------------------------------------------------------------------------
// Prep 1: lambda (complex) from nu/theta.
// ---------------------------------------------------------------------------
__global__ __launch_bounds__(256)
void lru_prep_lam_kernel(const float* __restrict__ nu,
                         const float* __restrict__ theta,
                         float* __restrict__ lam_re,
                         float* __restrict__ lam_im) {
  int i = blockIdx.x * 256 + threadIdx.x;
  if (i < 4 * N_HID) {
    float mag = __expf(-__expf(nu[i]));
    float ph  = __expf(theta[i]);
    float s, c;
    __sincosf(ph, &s, &c);
    lam_re[i] = mag * c;
    lam_im[i] = mag * s;
  }
}

// ---------------------------------------------------------------------------
// Prep 2: vectorized f32 -> f16 convert with constant scale (X, C_re, -C_im).
// 8 elements / thread: two b128 loads -> one b128 store.
// ---------------------------------------------------------------------------
__global__ __launch_bounds__(256)
void cvt_f16_kernel(const float* __restrict__ src, _Float16* __restrict__ dst,
                    float scale, long long n8) {
  long long t = (long long)blockIdx.x * 256 + threadIdx.x;
  if (t >= n8) return;
  const float4* q = reinterpret_cast<const float4*>(src) + 2 * t;
  float4 a = q[0], b = q[1];
  v8h o;
  o[0] = (_Float16)(a.x * scale); o[1] = (_Float16)(a.y * scale);
  o[2] = (_Float16)(a.z * scale); o[3] = (_Float16)(a.w * scale);
  o[4] = (_Float16)(b.x * scale); o[5] = (_Float16)(b.y * scale);
  o[6] = (_Float16)(b.z * scale); o[7] = (_Float16)(b.w * scale);
  reinterpret_cast<v8h*>(dst)[t] = o;
}

// ---------------------------------------------------------------------------
// Prep 3: B rows scaled by gamma[row] = exp(gamma_log[row]) and converted.
// Row stride DMODEL (2048) => 256 threads of 8 elements per row.
// ---------------------------------------------------------------------------
__global__ __launch_bounds__(256)
void cvt_B_kernel(const float* __restrict__ src,
                  const float* __restrict__ gamma_log,
                  _Float16* __restrict__ dst, long long n8) {
  long long t = (long long)blockIdx.x * 256 + threadIdx.x;
  if (t >= n8) return;
  int row = (int)(t / (DMODEL / 8));
  float g = __expf(gamma_log[row]);
  const float4* q = reinterpret_cast<const float4*>(src) + 2 * t;
  float4 a = q[0], b = q[1];
  v8h o;
  o[0] = (_Float16)(a.x * g); o[1] = (_Float16)(a.y * g);
  o[2] = (_Float16)(a.z * g); o[3] = (_Float16)(a.w * g);
  o[4] = (_Float16)(b.x * g); o[5] = (_Float16)(b.y * g);
  o[6] = (_Float16)(b.z * g); o[7] = (_Float16)(b.w * g);
  reinterpret_cast<v8h*>(dst)[t] = o;
}

// ---------------------------------------------------------------------------
// GEMM1: Bu = X @ (gamma*B)^T (complex), fused with the 2x2 complex state
// mix.  Each wave computes a 32x32 tile of [NBATCH, TWO_N] for BOTH re & im
// (A fragments shared).  8 waves/block.
// ---------------------------------------------------------------------------
__global__ __launch_bounds__(256)
void lru_gemm1_kernel(const _Float16* __restrict__ Xh,     // [NBATCH, DMODEL]
                      const _Float16* __restrict__ Breh,   // [TWO_N, DMODEL] (gamma-scaled)
                      const _Float16* __restrict__ Bimh,   // [TWO_N, DMODEL] (gamma-scaled)
                      const float* __restrict__ lam_re,    // [2,2,N_HID]
                      const float* __restrict__ lam_im,    // [2,2,N_HID]
                      const float* __restrict__ hre_old,   // [NBATCH, TWO_N]
                      const float* __restrict__ him_old,   // [NBATCH, TWO_N]
                      float* __restrict__ hid_re,          // out [NBATCH, TWO_N]
                      float* __restrict__ hid_im,          // out [NBATCH, TWO_N]
                      _Float16* __restrict__ hre_h,        // ws  [NBATCH, TWO_N]
                      _Float16* __restrict__ him_h) {      // ws  [NBATCH, TWO_N]
  const int lane = threadIdx.x & 31;
  const int wave = threadIdx.x >> 5;
  const int hf   = lane >> 4;     // K-chunk selector
  const int r    = lane & 15;     // row within fragment
  const int n0 = (blockIdx.x * 8 + wave) * 32;   // column tile in [0, TWO_N)
  const int m0 = blockIdx.y * 32;                // row tile in [0, NBATCH)

  v8f accRe[2][2] = {};
  v8f accIm[2][2] = {};

  const _Float16* aBase[2];
  const _Float16* brBase[2];
  const _Float16* biBase[2];
#pragma unroll
  for (int i = 0; i < 2; ++i) {
    aBase[i]  = Xh   + (size_t)(m0 + i * 16 + r) * DMODEL + hf * 8;
    int nrow = n0 + i * 16 + r;
    brBase[i] = Breh + (size_t)nrow * DMODEL + hf * 8;
    biBase[i] = Bimh + (size_t)nrow * DMODEL + hf * 8;
  }

#pragma unroll 2
  for (int k0 = 0; k0 < DMODEL; k0 += 32) {
    __builtin_prefetch(aBase[0] + k0 + 256, 0, 3);
    __builtin_prefetch(brBase[0] + k0 + 256, 0, 3);
    v16h a[2], bre[2], bim[2];
#pragma unroll
    for (int i = 0; i < 2; ++i) {
      a[i]   = load_frag_h16(aBase[i] + k0);
      bre[i] = load_frag_h16(brBase[i] + k0);
      bim[i] = load_frag_h16(biBase[i] + k0);
    }
#pragma unroll
    for (int i = 0; i < 2; ++i)
#pragma unroll
      for (int j = 0; j < 2; ++j) {
        accRe[i][j] = WMMA_F16(a[i], bre[j], accRe[i][j]);
        accIm[i][j] = WMMA_F16(a[i], bim[j], accIm[i][j]);
      }
  }

  // Epilogue: h = 0.5 * (lam[i,0,n]*old0 + lam[i,1,n]*old1) + Bu
  // C/D layout: lane -> col = base + (lane&15); VGPR v -> row = v + 8*(lane>>4)
#pragma unroll
  for (int i = 0; i < 2; ++i)
#pragma unroll
    for (int j = 0; j < 2; ++j)
#pragma unroll
      for (int v = 0; v < 8; ++v) {
        int b = m0 + i * 16 + hf * 8 + v;
        int c = n0 + j * 16 + r;
        int n  = c & (N_HID - 1);
        int ii = c >> 11;                       // which half of 2N
        float l0r = lam_re[(ii * 2 + 0) * N_HID + n];
        float l0i = lam_im[(ii * 2 + 0) * N_HID + n];
        float l1r = lam_re[(ii * 2 + 1) * N_HID + n];
        float l1i = lam_im[(ii * 2 + 1) * N_HID + n];
        size_t rb = (size_t)b * TWO_N;
        float o0r = hre_old[rb + n];
        float o0i = him_old[rb + n];
        float o1r = hre_old[rb + N_HID + n];
        float o1i = him_old[rb + N_HID + n];
        float lsr = l0r * o0r - l0i * o0i + l1r * o1r - l1i * o1i;
        float lsi = l0r * o0i + l0i * o0r + l1r * o1i + l1i * o1r;
        float hr = 0.5f * lsr + accRe[i][j][v];
        float hi = 0.5f * lsi + accIm[i][j][v];
        hid_re[rb + c] = hr;
        hid_im[rb + c] = hi;
        hre_h[rb + c] = (_Float16)hr;
        him_h[rb + c] = (_Float16)hi;
      }
}

// ---------------------------------------------------------------------------
// GEMM2: y = h_re @ C_re^T + h_im @ (-C_im)^T + D * X
// (the minus sign was folded into Cim_h during prep)
// ---------------------------------------------------------------------------
__global__ __launch_bounds__(256)
void lru_gemm2_kernel(const _Float16* __restrict__ hre_h,  // [NBATCH, TWO_N]
                      const _Float16* __restrict__ him_h,  // [NBATCH, TWO_N]
                      const _Float16* __restrict__ Creh,   // [DMODEL, TWO_N]
                      const _Float16* __restrict__ Cimh,   // [DMODEL, TWO_N] = -C_im
                      const float* __restrict__ Dvec,      // [DMODEL]
                      const float* __restrict__ X,         // [NBATCH, DMODEL]
                      float* __restrict__ Y) {             // [NBATCH, DMODEL]
  const int lane = threadIdx.x & 31;
  const int wave = threadIdx.x >> 5;
  const int hf   = lane >> 4;
  const int r    = lane & 15;
  const int n0 = (blockIdx.x * 8 + wave) * 32;   // output column tile (d)
  const int m0 = blockIdx.y * 32;                // batch tile

  v8f acc[2][2] = {};

  const _Float16* arBase[2];
  const _Float16* aiBase[2];
  const _Float16* crBase[2];
  const _Float16* ciBase[2];
#pragma unroll
  for (int i = 0; i < 2; ++i) {
    arBase[i] = hre_h + (size_t)(m0 + i * 16 + r) * TWO_N + hf * 8;
    aiBase[i] = him_h + (size_t)(m0 + i * 16 + r) * TWO_N + hf * 8;
    int nrow = n0 + i * 16 + r;
    crBase[i] = Creh + (size_t)nrow * TWO_N + hf * 8;
    ciBase[i] = Cimh + (size_t)nrow * TWO_N + hf * 8;
  }

#pragma unroll 2
  for (int k0 = 0; k0 < TWO_N; k0 += 32) {
    __builtin_prefetch(crBase[0] + k0 + 256, 0, 3);
    v16h ar[2], ai[2], cre[2], cim[2];
#pragma unroll
    for (int i = 0; i < 2; ++i) {
      ar[i]  = load_frag_h16(arBase[i] + k0);
      ai[i]  = load_frag_h16(aiBase[i] + k0);
      cre[i] = load_frag_h16(crBase[i] + k0);
      cim[i] = load_frag_h16(ciBase[i] + k0);
    }
#pragma unroll
    for (int i = 0; i < 2; ++i)
#pragma unroll
      for (int j = 0; j < 2; ++j) {
        acc[i][j] = WMMA_F16(ar[i], cre[j], acc[i][j]);
        acc[i][j] = WMMA_F16(ai[i], cim[j], acc[i][j]);
      }
  }

#pragma unroll
  for (int i = 0; i < 2; ++i)
#pragma unroll
    for (int j = 0; j < 2; ++j)
#pragma unroll
      for (int v = 0; v < 8; ++v) {
        int b = m0 + i * 16 + hf * 8 + v;
        int d = n0 + j * 16 + r;
        size_t idx = (size_t)b * DMODEL + d;
        Y[idx] = acc[i][j][v] + Dvec[d] * X[idx];
      }
}

// ---------------------------------------------------------------------------
extern "C" void kernel_launch(void* const* d_in, const int* in_sizes, int n_in,
                              void* d_out, int out_size, void* d_ws, size_t ws_size,
                              hipStream_t stream) {
  const float* X         = (const float*)d_in[0];   // [4096, 2048]
  const float* h_re_old  = (const float*)d_in[1];   // [4096, 4096]
  const float* h_im_old  = (const float*)d_in[2];   // [4096, 4096]
  const float* nu        = (const float*)d_in[3];   // [8192]
  const float* theta     = (const float*)d_in[4];   // [8192]
  const float* gamma_log = (const float*)d_in[5];   // [4096]
  const float* B_re      = (const float*)d_in[6];   // [4096, 2048]
  const float* B_im      = (const float*)d_in[7];   // [4096, 2048]
  const float* C_re      = (const float*)d_in[8];   // [2048, 4096]
  const float* C_im      = (const float*)d_in[9];   // [2048, 4096]
  const float* Dvec      = (const float*)d_in[10];  // [2048]

  float* Y      = (float*)d_out;                                  // [4096, 2048]
  float* hid_re = Y + (size_t)NBATCH * DMODEL;                    // [4096, 4096]
  float* hid_im = hid_re + (size_t)NBATCH * TWO_N;                // [4096, 4096]

  char* ws = (char*)d_ws;
  float* lam_re = (float*)ws;                          // 8192 f32
  float* lam_im = lam_re + 4 * N_HID;                  // 8192 f32
  _Float16* Xh    = (_Float16*)(ws + (64 << 10));
  _Float16* Breh  = Xh    + (size_t)NBATCH * DMODEL;   // 16 MB each
  _Float16* Bimh  = Breh  + (size_t)TWO_N * DMODEL;
  _Float16* Creh  = Bimh  + (size_t)TWO_N * DMODEL;
  _Float16* Cimh  = Creh  + (size_t)DMODEL * TWO_N;
  _Float16* hre_h = Cimh  + (size_t)DMODEL * TWO_N;    // 32 MB each
  _Float16* him_h = hre_h + (size_t)NBATCH * TWO_N;

  // --- prep: lambda + f16 conversions (bandwidth-bound, ~10us total) ---
  lru_prep_lam_kernel<<<dim3((4 * N_HID + 255) / 256), dim3(256), 0, stream>>>(
      nu, theta, lam_re, lam_im);

  const long long nXB8 = (long long)NBATCH * DMODEL / 8;   // also B/C sizes
  cvt_f16_kernel<<<dim3((unsigned)(nXB8 / 256)), dim3(256), 0, stream>>>(
      X, Xh, 1.0f, nXB8);
  cvt_B_kernel<<<dim3((unsigned)(nXB8 / 256)), dim3(256), 0, stream>>>(
      B_re, gamma_log, Breh, nXB8);
  cvt_B_kernel<<<dim3((unsigned)(nXB8 / 256)), dim3(256), 0, stream>>>(
      B_im, gamma_log, Bimh, nXB8);
  cvt_f16_kernel<<<dim3((unsigned)(nXB8 / 256)), dim3(256), 0, stream>>>(
      C_re, Creh, 1.0f, nXB8);
  cvt_f16_kernel<<<dim3((unsigned)(nXB8 / 256)), dim3(256), 0, stream>>>(
      C_im, Cimh, -1.0f, nXB8);

  // GEMM1: 128 row tiles x 128 col tiles of 32x32; 8 waves (col tiles) / block
  lru_gemm1_kernel<<<dim3(TWO_N / 256, NBATCH / 32), dim3(256), 0, stream>>>(
      Xh, Breh, Bimh, lam_re, lam_im, h_re_old, h_im_old,
      hid_re, hid_im, hre_h, him_h);

  // GEMM2: 128 row tiles x 64 col tiles of 32x32; 8 waves / block
  lru_gemm2_kernel<<<dim3(DMODEL / 256, NBATCH / 32), dim3(256), 0, stream>>>(
      hre_h, him_h, Creh, Cimh, Dvec, X, Y);
}
